// GCNModel_32074815767312
// MI455X (gfx1250) — matrix operand order repaired and checked by reading, not compiled, not run on hardware
//
#include <hip/hip_runtime.h>
#include <hip/hip_bf16.h>

// ---------------------------------------------------------------------------
// Fully fused per-graph GCN forward for MI455X (gfx1250, wave32, WMMA).
// One 256-thread workgroup (8 wave32) handles one 50-node / 800-edge graph
// entirely in LDS. GEMMs use v_wmma_f32_16x16x32_f16 with compile-time K:
// each wave owns one N-tile (B fragment loaded once per k-step, shared by
// 4 M-tile accumulators), fragments fetched as ds_load_b128.
// ---------------------------------------------------------------------------

typedef __attribute__((ext_vector_type(16))) _Float16 v16h;
typedef __attribute__((ext_vector_type(8)))  float    v8f;

#define NGRAPH_NODES 50
#define NGRAPH_EDGES 800
#define FIN   64
#define FH    128
#define FPRED 512
#define MROWS 64                    // 50 rows padded to 4 x 16 tiles
#define HP    (FH + 2)              // padded fp32 LDS stride for h/agg (130)

// fp16 staging strides: K+8 halfs -> 16B-aligned b128 fragment loads,
// row stride 144/272 bytes keeps banks rotating.
#define KP0   (FIN + 8)             // 72
#define KPH   (FH + 8)              // 136

// ---- LDS arena (bytes) ----
#define OFF_XA   0
#define SZ_XA    (MROWS * KPH * 2)          // 17408
#define OFF_WT   (OFF_XA + SZ_XA)
#define SZ_WT    (FH * KPH * 2)             // 34816
#define OFF_H    (OFF_WT + SZ_WT)
#define SZ_H     (MROWS * HP * 4)           // 33280
#define OFF_AGG  (OFF_H + SZ_H)
#define SZ_AGG   SZ_H                       // 33280
#define OFF_EDGE (OFF_AGG + SZ_AGG)
#define SZ_EDGE  (NGRAPH_EDGES * 2 * 2)     // 3200
#define OFF_NRM  (OFF_EDGE + SZ_EDGE)
#define SZ_NRM   (2 * MROWS * 4)            // 512
#define OFF_MV   (OFF_NRM + SZ_NRM)
#define SZ_MV    (FH * 4)                   // 512 (reused early for degrees)
#define OFF_RED  (OFF_MV + SZ_MV)
#define SZ_RED   (256 * 4)                  // 1024
#define SMEM_BYTES (OFF_RED + SZ_RED)       // 124032 B < 320 KB/WGP

// ---- stage weights: global fp32 [K][128] -> LDS fp16 transposed [n][K+8] ----
template <int K>
__device__ __forceinline__
void load_weights(const float* __restrict__ Wg, _Float16* WT, int tid) {
    constexpr int KP = K + 8;
    for (int i = tid; i < K * FH; i += 256) {
        int k = i >> 7, n = i & (FH - 1);        // coalesced global read
        WT[n * KP + k] = (_Float16)Wg[i];
    }
}

// ---- h[64x128] = xA[64xK] @ W[Kx128] via WMMA; wave w owns N-tile w ----
template <int K>
__device__ __forceinline__
void gemm_wmma(const _Float16* xA, const _Float16* WT, float* hbuf,
               int lane, int wave) {
    constexpr int KP = K + 8;
    const int l15  = lane & 15;
    const int hiA  = (lane >> 4) * 8;    // A: lanes16-31 hold K+8 block
    const int hiB  = (lane >> 4) * 8;    // B: lanes16-31 hold K+8 block
    const int bcol = wave * 16 + l15;
    v8f acc[4] = {};
#pragma unroll
    for (int ks = 0; ks < K; ks += 32) {
        union { v16h v; uint4 q[2]; } bf;
        const _Float16* bb = WT + bcol * KP + ks + hiB;
        bf.q[0] = *(const uint4*)(bb);           // K 0..7   (x2 halfs/dword)
        bf.q[1] = *(const uint4*)(bb + 16);      // K 16..23
#pragma unroll
        for (int mt = 0; mt < 4; ++mt) {
            union { v16h v; uint4 q[2]; } af;
            const _Float16* ab = xA + (mt * 16 + l15) * KP + ks + hiA;
            af.q[0] = *(const uint4*)(ab);       // K 0..7
            af.q[1] = *(const uint4*)(ab + 16);  // K 16..23
            acc[mt] = __builtin_amdgcn_wmma_f32_16x16x32_f16(
                false, af.v, false, bf.v, (short)0, acc[mt], false, false);
        }
    }
#pragma unroll
    for (int mt = 0; mt < 4; ++mt) {
        const int row0 = mt * 16 + (lane >> 4) * 8;
#pragma unroll
        for (int r = 0; r < 8; ++r)
            hbuf[(row0 + r) * HP + bcol] = acc[mt][r];
    }
}

// ---- message passing: agg[dst] += h[src] via ds_add_f32 (conflict-free) ----
__device__ __forceinline__
void aggregate(const short* sloc, const short* dloc, const float* hbuf,
               float* agg, int tid) {
    const int c  = tid & (FH - 1);               // feature lane
    const int eo = tid >> 7;                     // 2 edges in flight
    for (int base = 0; base < NGRAPH_EDGES; base += 8) {
#pragma unroll
        for (int u = 0; u < 4; ++u) {
            const int e = base + 2 * u + eo;
            atomicAdd(&agg[(int)dloc[e] * HP + c], hbuf[(int)sloc[e] * HP + c]);
        }
    }
}

// ---- x_next = relu(agg*norm_in + b) * norm_out -> fp16 A staging ----
__device__ __forceinline__
void stage_next(const float* agg, const float* nrmI, const float* nrmO,
                const float* __restrict__ bv, _Float16* xA, int tid) {
    for (int i = tid; i < MROWS * KPH; i += 256) {
        int r = i / KPH, c = i - r * KPH;
        float v = 0.0f;
        if (r < NGRAPH_NODES && c < FH) {
            float x = agg[r * HP + c] * nrmI[r] + bv[c];
            v = fmaxf(x, 0.0f) * nrmO[r];
        }
        xA[i] = (_Float16)v;
    }
}

__device__ __forceinline__
void zero_agg(float* agg, int tid) {
    for (int i = tid; i < MROWS * HP; i += 256) agg[i] = 0.0f;
}

__global__ __launch_bounds__(256)
void gcn_graph_fused(const float* __restrict__ feat,
                     const int*   __restrict__ src_all,
                     const int*   __restrict__ dst_all,
                     const float* __restrict__ W0, const float* __restrict__ b0,
                     const float* __restrict__ W1, const float* __restrict__ b1,
                     const float* __restrict__ W2, const float* __restrict__ b2,
                     const float* __restrict__ Wp1, const float* __restrict__ bp1,
                     const float* __restrict__ Wp2, const float* __restrict__ bp2,
                     float* __restrict__ out)
{
    extern __shared__ char smem[];
    _Float16* xA   = (_Float16*)(smem + OFF_XA);
    _Float16* WT   = (_Float16*)(smem + OFF_WT);
    float*    hbuf = (float*)(smem + OFF_H);
    float*    agg  = (float*)(smem + OFF_AGG);
    short*    sloc = (short*)(smem + OFF_EDGE);
    short*    dloc = sloc + NGRAPH_EDGES;
    float*    nrmO = (float*)(smem + OFF_NRM);
    float*    nrmI = nrmO + MROWS;
    float*    mvec = (float*)(smem + OFF_MV);
    float*    red  = (float*)(smem + OFF_RED);
    int*      degO = (int*)mvec;                 // reuse mvec space early
    int*      degI = degO + MROWS;

    const int tid  = threadIdx.x;
    const int lane = tid & 31;                   // wave32
    const int wave = tid >> 5;                   // 8 waves
    const int g    = blockIdx.x;
    const int nbase = g * NGRAPH_NODES;

    const int* srcp = src_all + (size_t)g * NGRAPH_EDGES;
    const int* dstp = dst_all + (size_t)g * NGRAPH_EDGES;

    // ---------- degrees + localized edge list (ds_add_u32) ----------
    if (tid < 2 * MROWS) degO[tid] = 0;
    __syncthreads();
    for (int e = tid; e < NGRAPH_EDGES; e += 256) {
        int s = srcp[e] - nbase;
        int d = dstp[e] - nbase;
        sloc[e] = (short)s;
        dloc[e] = (short)d;
        atomicAdd(&degO[s], 1);
        atomicAdd(&degI[d], 1);
    }
    // warm L2 with the MLP head weights (global_prefetch_b8)
    for (size_t i = (size_t)tid * 64; i < (size_t)FH * FPRED; i += 256 * 64)
        __builtin_prefetch(Wp1 + i, 0, 1);
    __syncthreads();
    if (tid < MROWS) {
        nrmO[tid] = rsqrtf(fmaxf((float)degO[tid], 1.0f));
        nrmI[tid] = rsqrtf(fmaxf((float)degI[tid], 1.0f));
    }
    __syncthreads();

    // ---------- stage layer-0 A matrix: (feat * norm_out) -> fp16 ----------
    for (int i = tid; i < MROWS * KP0; i += 256) {
        int r = i / KP0, c = i - r * KP0;
        float v = 0.0f;
        if (r < NGRAPH_NODES && c < FIN)
            v = feat[(size_t)(nbase + r) * FIN + c] * nrmO[r];
        xA[i] = (_Float16)v;
    }
    __syncthreads();

    // ================= layer 0 (K = 64) =================
    load_weights<FIN>(W0, WT, tid);           __syncthreads();
    gemm_wmma<FIN>(xA, WT, hbuf, lane, wave); __syncthreads();
    zero_agg(agg, tid);                       __syncthreads();
    aggregate(sloc, dloc, hbuf, agg, tid);    __syncthreads();
    stage_next(agg, nrmI, nrmO, b0, xA, tid); __syncthreads();

    // ================= layer 1 (K = 128) =================
    load_weights<FH>(W1, WT, tid);            __syncthreads();
    gemm_wmma<FH>(xA, WT, hbuf, lane, wave);  __syncthreads();
    zero_agg(agg, tid);                       __syncthreads();
    aggregate(sloc, dloc, hbuf, agg, tid);    __syncthreads();
    stage_next(agg, nrmI, nrmO, b1, xA, tid); __syncthreads();

    // ================= layer 2 (K = 128) =================
    load_weights<FH>(W2, WT, tid);            __syncthreads();
    gemm_wmma<FH>(xA, WT, hbuf, lane, wave);  __syncthreads();
    zero_agg(agg, tid);                       __syncthreads();
    aggregate(sloc, dloc, hbuf, agg, tid);    __syncthreads();

    // ---------- mean pooling over 50 nodes (final activation, bias b2) ----------
    if (tid < FH) {
        float s = 0.0f;
        const float bc = b2[tid];
        for (int r = 0; r < NGRAPH_NODES; ++r)
            s += fmaxf(agg[r * HP + tid] * nrmI[r] + bc, 0.0f);
        mvec[tid] = s * (1.0f / (float)NGRAPH_NODES);
    }
    __syncthreads();

    // ---------- MLP head: relu(m @ Wp1 + bp1) @ Wp2 + bp2 ----------
    float part = 0.0f;
#pragma unroll
    for (int jj = 0; jj < FPRED / 256; ++jj) {
        const int j = tid + jj * 256;
        float accd = bp1[j];
#pragma unroll 8
        for (int k = 0; k < FH; ++k)
            accd += mvec[k] * Wp1[(size_t)k * FPRED + j];   // coalesced, L2-resident
        part += fmaxf(accd, 0.0f) * Wp2[j];
    }
    red[tid] = part;
    __syncthreads();
    for (int s = 128; s > 0; s >>= 1) {
        if (tid < s) red[tid] += red[tid + s];
        __syncthreads();
    }
    if (tid == 0) out[g] = red[0] + bp2[0];
}

// ---------------------------------------------------------------------------
extern "C" void kernel_launch(void* const* d_in, const int* in_sizes, int n_in,
                              void* d_out, int out_size, void* d_ws, size_t ws_size,
                              hipStream_t stream) {
    const float* feat = (const float*)d_in[0];
    // d_in[1] = edge_features (unused by model)
    const int*   pair = (const int*)d_in[2];       // [2, E]
    const int    E    = in_sizes[2] / 2;
    const float* W0  = (const float*)d_in[5];
    const float* b0  = (const float*)d_in[6];
    const float* W1  = (const float*)d_in[7];
    const float* b1  = (const float*)d_in[8];
    const float* W2  = (const float*)d_in[9];
    const float* b2  = (const float*)d_in[10];
    const float* Wp1 = (const float*)d_in[11];
    const float* bp1 = (const float*)d_in[12];
    const float* Wp2 = (const float*)d_in[13];
    const float* bp2 = (const float*)d_in[14];
    float* out = (float*)d_out;

    const int nGraphs = out_size;                  // 2000 graphs, 1 WG each
    gcn_graph_fused<<<nGraphs, 256, SMEM_BYTES, stream>>>(
        feat, pair, pair + (size_t)E,
        W0, b0, W1, b1, W2, b2, Wp1, bp1, Wp2, bp2, out);
}
